// GAT_18202071400736
// MI455X (gfx1250) — matrix operand order repaired and checked by reading, not compiled, not run on hardware
//
#include <hip/hip_runtime.h>
#include <hip/hip_bf16.h>
#include <math.h>

// GAT 3-layer forward for MI455X (gfx1250, wave32).
// GEMMs via v_wmma_f32_16x16x32_bf16 on pre-converted bf16 operands;
// softmax-aggregation via L2-resident gathers + f32 atomics.

typedef __attribute__((ext_vector_type(16))) __bf16 v16bf;
typedef __attribute__((ext_vector_type(8)))  __bf16 v8bf;
typedef __attribute__((ext_vector_type(4)))  __bf16 v4bf;
typedef __attribute__((ext_vector_type(8)))  float  v8f;
typedef __attribute__((ext_vector_type(4)))  float  v4f;

#define KD     128   // feature width of every layer input/output (H*C)
#define NHEADS 4
#define CH     32

// ---------------------------------------------------------------------------
// fp32 -> bf16 bulk convert (done once per layer for X and W)
// ---------------------------------------------------------------------------
__global__ void cvt_f32_bf16(const float* __restrict__ in,
                             __bf16* __restrict__ out, int n4)
{
  int i = blockIdx.x * blockDim.x + threadIdx.x;   // one float4 per thread
  if (i >= n4) return;
  v4f v = *(const v4f*)(in + (size_t)i * 4);
  v4bf o = { (__bf16)v[0], (__bf16)v[1], (__bf16)v[2], (__bf16)v[3] };
  *(v4bf*)(out + (size_t)i * 4) = o;
}

// ---------------------------------------------------------------------------
// h[N,128] = X[N,128] @ W[128,128]^T   (h[n,o] = sum_k X[n,k]*W[o,k])
// One block = 16 rows of output. 8 waves, wave w owns cols [16w,16w+16).
// A layout (16-bit 16x32): lane%16 = M; elem j -> K = kk+half*16+j
//                          (half = lane/16)
// B layout (32x16):        lane%16 = N; elem j -> K = kk+half*16+j
// D layout (16x16 f32):    lane%16 = N; vgpr r -> M = r + 8*half
// ---------------------------------------------------------------------------
__global__ __launch_bounds__(256) void gat_gemm_wmma(
    const __bf16* __restrict__ Xb, const __bf16* __restrict__ Wb,
    float* __restrict__ H)
{
  const int wave = threadIdx.x >> 5;
  const int lane = threadIdx.x & 31;
  const int half = lane >> 4;
  const int lm   = lane & 15;
  const int row  = blockIdx.x * 16 + lm;   // A row handled by this lane
  const int col  = wave * 16 + lm;         // B/D column handled by this lane

  const __bf16* __restrict__ arow = Xb + (size_t)row * KD;
  const __bf16* __restrict__ wrow = Wb + (size_t)col * KD;  // B[k][col] = W[col][k]

  v8f acc = {};
#pragma unroll
  for (int kk = 0; kk < KD; kk += 32) {
    v16bf a = *(const v16bf*)(arow + kk + half * 16);       // K = kk+half*16 .. +15
    v16bf b = *(const v16bf*)(wrow + kk + half * 16);       // K = kk+half*16 .. +15
    acc = __builtin_amdgcn_wmma_f32_16x16x32_bf16(
        /*neg_a=*/false, a, /*neg_b=*/false, b,
        /*c_mod=*/(short)0, acc, /*reuse_a=*/false, /*reuse_b=*/false);
  }

  float* __restrict__ hp = H + (size_t)(blockIdx.x * 16 + 8 * half) * KD + col;
#pragma unroll
  for (int r = 0; r < 8; ++r)
    hp[(size_t)r * KD] = acc[r];
}

// ---------------------------------------------------------------------------
// Per-(node,head) attention logits: als = <h[n,h,:], a_src[h,:]>, ald likewise
// ---------------------------------------------------------------------------
__global__ void gat_scores(const float* __restrict__ H,
                           const float* __restrict__ a_src,
                           const float* __restrict__ a_dst,
                           float* __restrict__ als, float* __restrict__ ald,
                           int nh)
{
  int i = blockIdx.x * blockDim.x + threadIdx.x;   // node*NHEADS + head
  if (i >= nh) return;
  int node = i >> 2, head = i & 3;
  const float* hp = H + (size_t)node * KD + head * CH;
  const float* as = a_src + head * CH;
  const float* ad = a_dst + head * CH;
  float s = 0.f, d = 0.f;
#pragma unroll
  for (int c = 0; c < CH; ++c) { float v = hp[c]; s += v * as[c]; d += v * ad[c]; }
  als[i] = s; ald[i] = d;
}

// order-preserving float<->uint mapping for atomic max
__device__ __forceinline__ unsigned fmap(float f) {
  unsigned u = __float_as_uint(f);
  return (u & 0x80000000u) ? ~u : (u | 0x80000000u);
}
__device__ __forceinline__ float funmap(unsigned m) {
  return __uint_as_float((m & 0x80000000u) ? (m & 0x7fffffffu) : ~m);
}
__device__ __forceinline__ float lrelu(float s) {
  return s > 0.f ? s : 0.2f * s;
}

// ---------------------------------------------------------------------------
// Segment max over edges sharing a destination. One thread per edge,
// float4 over the 4 heads (contiguous), uint-mapped atomic max.
// ---------------------------------------------------------------------------
__global__ void gat_edge_max(const int* __restrict__ src, const int* __restrict__ dst,
                             const float* __restrict__ als, const float* __restrict__ ald,
                             unsigned* __restrict__ mu, int E)
{
  int e = blockIdx.x * blockDim.x + threadIdx.x;
  if (e >= E) return;
  int s = src[e], d = dst[e];
  v4f a = *(const v4f*)(als + (size_t)s * NHEADS);
  v4f b = *(const v4f*)(ald + (size_t)d * NHEADS);
#pragma unroll
  for (int h = 0; h < NHEADS; ++h)
    atomicMax(mu + (size_t)d * NHEADS + h, fmap(lrelu(a[h] + b[h])));
}

// ---------------------------------------------------------------------------
// exp(s - max) per edge-head; accumulate denominator per (dst,head)
// ---------------------------------------------------------------------------
__global__ void gat_edge_expsum(const int* __restrict__ src, const int* __restrict__ dst,
                                const float* __restrict__ als, const float* __restrict__ ald,
                                const unsigned* __restrict__ mu,
                                float* __restrict__ ev, float* __restrict__ den, int E)
{
  int e = blockIdx.x * blockDim.x + threadIdx.x;
  if (e >= E) return;
  int s = src[e], d = dst[e];
  v4f a = *(const v4f*)(als + (size_t)s * NHEADS);
  v4f b = *(const v4f*)(ald + (size_t)d * NHEADS);
  v4f ex;
#pragma unroll
  for (int h = 0; h < NHEADS; ++h) {
    float m = funmap(mu[(size_t)d * NHEADS + h]);
    ex[h] = __expf(lrelu(a[h] + b[h]) - m);
    atomicAdd(den + (size_t)d * NHEADS + h, ex[h]);
  }
  *(v4f*)(ev + (size_t)e * NHEADS) = ex;
}

// ---------------------------------------------------------------------------
// out[dst] += alpha * h[src] — 32 threads per edge, float4 features each.
// h and out are L2-resident (25.6 MB each): gather+atomics run at L2 speed.
// ---------------------------------------------------------------------------
__global__ __launch_bounds__(256) void gat_aggregate(
    const int* __restrict__ src, const int* __restrict__ dst,
    const float* __restrict__ H, const float* __restrict__ ev,
    const float* __restrict__ den, float* __restrict__ out, int E)
{
  int e = blockIdx.x * 8 + (threadIdx.x >> 5);
  if (e >= E) return;
  int t = (threadIdx.x & 31) * 4;   // feature base; head = t/32
  int h = t >> 5;
  int s = src[e], d = dst[e];
  float alpha = ev[(size_t)e * NHEADS + h] / (den[(size_t)d * NHEADS + h] + 1e-16f);
  v4f hv = *(const v4f*)(H + (size_t)s * KD + t);
  float* op = out + (size_t)d * KD + t;
  atomicAdd(op + 0, alpha * hv[0]);
  atomicAdd(op + 1, alpha * hv[1]);
  atomicAdd(op + 2, alpha * hv[2]);
  atomicAdd(op + 3, alpha * hv[3]);
}

// ---------------------------------------------------------------------------
// bias add (+ELU for non-final layers), write to next-layer input / d_out
// ---------------------------------------------------------------------------
__global__ void gat_finalize(const float* __restrict__ agg, const float* __restrict__ b,
                             float* __restrict__ out, int n, int apply_elu)
{
  int i = blockIdx.x * blockDim.x + threadIdx.x;
  if (i >= n) return;
  float v = agg[i] + b[i & (KD - 1)];
  if (apply_elu && v <= 0.f) v = expm1f(v);
  out[i] = v;
}

extern "C" void kernel_launch(void* const* d_in, const int* in_sizes, int n_in,
                              void* d_out, int out_size, void* d_ws, size_t ws_size,
                              hipStream_t stream)
{
  const float* x   = (const float*)d_in[0];
  const int*   src = (const int*)d_in[1];
  const int*   dst = (const int*)d_in[2];
  const int N = in_sizes[0] / KD;      // 50000 (divisible by 16)
  const int E = in_sizes[1];           // 850000

  // workspace carve-up
  float* ws  = (float*)d_ws;
  float* h   = ws;                               // N*128 f32
  float* cur = h   + (size_t)N * KD;             // N*128 f32 (layer ping buffer)
  float* agg = cur + (size_t)N * KD;             // N*128 f32 (aggregation accum)
  float* als = agg + (size_t)N * KD;             // N*4
  float* ald = als + (size_t)N * NHEADS;         // N*4
  float* den = ald + (size_t)N * NHEADS;         // N*4
  float* ev  = den + (size_t)N * NHEADS;         // E*4
  unsigned* mu = (unsigned*)(ev + (size_t)E * NHEADS);          // N*4 u32
  __bf16* xb = (__bf16*)(mu + (size_t)N * NHEADS);              // N*128 bf16
  __bf16* wb = xb + (size_t)N * KD;                             // 128*128 bf16

  const int nh = N * NHEADS;
  const float* in_ptr = x;

  for (int l = 0; l < 3; ++l) {
    const float* W  = (const float*)d_in[3 + l * 4 + 0];
    const float* as = (const float*)d_in[3 + l * 4 + 1];
    const float* ad = (const float*)d_in[3 + l * 4 + 2];
    const float* b  = (const float*)d_in[3 + l * 4 + 3];

    hipMemsetAsync(agg, 0, (size_t)N * KD * sizeof(float), stream);
    hipMemsetAsync(den, 0, (size_t)N * NHEADS * sizeof(float), stream);
    hipMemsetAsync(mu,  0, (size_t)N * NHEADS * sizeof(unsigned), stream);

    // one-time per-layer bf16 conversion of GEMM operands
    int nx4 = (N * KD) / 4, nw4 = (KD * KD) / 4;
    cvt_f32_bf16<<<(nx4 + 255) / 256, 256, 0, stream>>>(in_ptr, xb, nx4);
    cvt_f32_bf16<<<(nw4 + 255) / 256, 256, 0, stream>>>(W, wb, nw4);

    gat_gemm_wmma<<<N / 16, 256, 0, stream>>>(xb, wb, h);
    gat_scores<<<(nh + 255) / 256, 256, 0, stream>>>(h, as, ad, als, ald, nh);
    gat_edge_max<<<(E + 255) / 256, 256, 0, stream>>>(src, dst, als, ald, mu, E);
    gat_edge_expsum<<<(E + 255) / 256, 256, 0, stream>>>(src, dst, als, ald, mu, ev, den, E);
    gat_aggregate<<<(E + 7) / 8, 256, 0, stream>>>(src, dst, h, ev, den, agg, E);

    float* outp = (l == 2) ? (float*)d_out : cur;
    gat_finalize<<<(N * KD + 255) / 256, 256, 0, stream>>>(agg, b, outp, N * KD, l < 2);
    in_ptr = cur;
  }
}
